// LayerNormLSTMCell_48335561949441
// MI455X (gfx1250) — matrix operand order repaired and checked by reading, not compiled, not run on hardware
//
#include <hip/hip_runtime.h>
#include <hip/hip_bf16.h>
#include <math.h>

typedef __attribute__((ext_vector_type(16))) __bf16 v16bf;
typedef __attribute__((ext_vector_type(8)))  float  v8f;

#define NB 4096
#define NI 1024
#define NH 1024
#define EPSV 1e-5f
#define FGB  (-1.0f)
#define NT   16          // tiles per wave per matrix (4 per gate group)
#define XI_STRIDE 132    // padded per-thread LDS slot in floats (132%64=4 -> spread banks)

__device__ __forceinline__ float sigmoid_f(float x) {
    return 1.0f / (1.0f + __expf(-x));
}

struct Raw4 { float4 a, b, c, d; };

// A operand raw: 16x32, K order used by this code (same as B) -> lane half 0: K 0..15, half 1: K 16..31 (contiguous)
__device__ __forceinline__ Raw4 load_a_raw(const float* __restrict__ p, int off) {
    const float4* p0 = (const float4*)(p + off);
    const float4* p1 = (const float4*)(p + off + 8);
    Raw4 r; r.a = p0[0]; r.b = p0[1]; r.c = p1[0]; r.d = p1[1];
    return r;
}

// B operand raw: 32x16, K order used by this code (same as A) -> lane half 0: K 0..15, half 1: K 16..31 (contiguous)
__device__ __forceinline__ Raw4 load_b_raw(const float* __restrict__ p, int off) {
    const float4* q = (const float4*)(p + off);
    Raw4 r; r.a = q[0]; r.b = q[1]; r.c = q[2]; r.d = q[3];
    return r;
}

__device__ __forceinline__ v16bf cvt16(const Raw4& x) {
    v16bf r;
    r[0]  = (__bf16)x.a.x; r[1]  = (__bf16)x.a.y; r[2]  = (__bf16)x.a.z; r[3]  = (__bf16)x.a.w;
    r[4]  = (__bf16)x.b.x; r[5]  = (__bf16)x.b.y; r[6]  = (__bf16)x.b.z; r[7]  = (__bf16)x.b.w;
    r[8]  = (__bf16)x.c.x; r[9]  = (__bf16)x.c.y; r[10] = (__bf16)x.c.z; r[11] = (__bf16)x.c.w;
    r[12] = (__bf16)x.d.x; r[13] = (__bf16)x.d.y; r[14] = (__bf16)x.d.z; r[15] = (__bf16)x.d.w;
    return r;
}

// One GEMM (act @ W^T + bias) over this workgroup's 16 rows and this wave's
// 16 column-tiles, followed by the 4096-wide row LayerNorm. Result (normalized)
// is left in acc[]. Contains one __syncthreads().
__device__ __forceinline__ void gemm_ln_phase(
    const float* __restrict__ act, const int aoff0,
    const float* __restrict__ w,
    const float* __restrict__ bias,
    const float* __restrict__ lnw, const float* __restrict__ lnb,
    float* __restrict__ s_sum, float* __restrict__ s_sq,
    const int wave, const int l15, const int khalf,
    v8f acc[NT])
{
#pragma unroll
    for (int t = 0; t < NT; ++t) acc[t] = {};

    // Wave-uniform scalar bases: gates 0,1 from wb0; gates 2,3 from wb1 (+8MB).
    const float* wb0 = w + (size_t)(wave * 64) * NI;
    const float* wb1 = w + (size_t)(wave * 64 + 2048) * NI;
    const int boff_l = l15 * NI;

    for (int k0 = 0; k0 < NI; k0 += 32) {
        const int kba = k0 + khalf * 16;   // A layout K split
        const int kbb = k0 + khalf * 16;   // B layout K split
        const int bo  = boff_l + kbb;

        v16bf a_m = cvt16(load_a_raw(act, aoff0 + kba));

        // 16 tiles; scheduler hoists/pipelines the b128 loads on its own.
#pragma unroll
        for (int t = 0; t < NT; ++t) {
            const int g = t >> 2, s = t & 3;
            const float* base = (g < 2) ? wb0 : wb1;
            const int cst = (((g & 1) * 1024) + s * 16) * NI;  // <= 4.4MB, folds to IOFFSET
            v16bf bm = cvt16(load_b_raw(base, bo + cst));
            acc[t] = __builtin_amdgcn_wmma_f32_16x16x32_bf16(
                false, a_m, false, bm, (short)0, acc[t], false, false);
        }
    }

    // bias add + per-row partial sums
    float psum[8], psq[8];
#pragma unroll
    for (int v = 0; v < 8; ++v) { psum[v] = 0.f; psq[v] = 0.f; }
#pragma unroll
    for (int t = 0; t < NT; ++t) {
        const int col = (t >> 2) * 1024 + wave * 64 + (t & 3) * 16 + l15;
        const float bb = bias[col];
#pragma unroll
        for (int v = 0; v < 8; ++v) {
            const float x = acc[t][v] + bb; acc[t][v] = x;
            psum[v] += x; psq[v] += x * x;
        }
    }
#pragma unroll
    for (int v = 0; v < 8; ++v) {
#pragma unroll
        for (int off = 1; off < 16; off <<= 1) {
            psum[v] += __shfl_xor(psum[v], off, 32);
            psq[v]  += __shfl_xor(psq[v],  off, 32);
        }
        if (l15 == 0) {
            const int r = v + khalf * 8;
            atomicAdd(&s_sum[r], psum[v]);
            atomicAdd(&s_sq[r],  psq[v]);
        }
    }
    __syncthreads();

    // unbiased std (ddof=1), n = 4096
    float mean[8], inv[8];
    const float nG = 4096.0f;
#pragma unroll
    for (int v = 0; v < 8; ++v) {
        const int r = v + khalf * 8;
        const float s = s_sum[r], q = s_sq[r];
        const float m = s / nG;
        const float var = fmaxf(0.f, (q - s * m) / (nG - 1.0f));
        mean[v] = m; inv[v] = 1.0f / (sqrtf(var) + EPSV);
    }
#pragma unroll
    for (int t = 0; t < NT; ++t) {
        const int col = (t >> 2) * 1024 + wave * 64 + (t & 3) * 16 + l15;
        const float wl = lnw[col], bl = lnb[col];
#pragma unroll
        for (int v = 0; v < 8; ++v)
            acc[t][v] = (acc[t][v] - mean[v]) * inv[v] * wl + bl;
    }
}

extern "C" __global__ __launch_bounds__(512, 1)
void lnlstm_fused_kernel(const float* __restrict__ xin,  const float* __restrict__ hx,
                         const float* __restrict__ cx,
                         const float* __restrict__ w_i,  const float* __restrict__ b_i,
                         const float* __restrict__ w_h,  const float* __restrict__ b_h,
                         const float* __restrict__ lnw_i, const float* __restrict__ lnb_i,
                         const float* __restrict__ lnw_h, const float* __restrict__ lnb_h,
                         const float* __restrict__ lnw_o, const float* __restrict__ lnb_o,
                         float* __restrict__ out_h, float* __restrict__ out_c)
{
    __shared__ float s_sum_i[16], s_sq_i[16];
    __shared__ float s_sum_h[16], s_sq_h[16];
    __shared__ float s_sum_c[16], s_sq_c[16];
    extern __shared__ float xi_buf[];   // 512 * XI_STRIDE floats

    const int tid   = threadIdx.x;
    const int lane  = tid & 31;
    const int wave  = __builtin_amdgcn_readfirstlane(tid >> 5);  // uniform -> SGPR bases
    const int l15   = lane & 15;
    const int khalf = lane >> 4;
    const int rb    = blockIdx.x * 16;

    if (tid < 16) {
        s_sum_i[tid] = 0.f; s_sq_i[tid] = 0.f;
        s_sum_h[tid] = 0.f; s_sq_h[tid] = 0.f;
        s_sum_c[tid] = 0.f; s_sq_c[tid] = 0.f;
    }
    __syncthreads();

    v8f acc[NT];

    // ---- Pass A: i2h GEMM + LN; stash normalized xi in this thread's LDS slot ----
    gemm_ln_phase(xin, (rb + l15) * NI, w_i, b_i, lnw_i, lnb_i,
                  s_sum_i, s_sq_i, wave, l15, khalf, acc);

    float* slot = &xi_buf[(size_t)tid * XI_STRIDE];
#pragma unroll
    for (int t = 0; t < NT; ++t) {
        float4 lo = make_float4(acc[t][0], acc[t][1], acc[t][2], acc[t][3]);
        float4 hi = make_float4(acc[t][4], acc[t][5], acc[t][6], acc[t][7]);
        *(float4*)&slot[t * 8 + 0] = lo;
        *(float4*)&slot[t * 8 + 4] = hi;
    }

    // ---- Pass B: h2h GEMM + LN (reuses acc registers) ----
    gemm_ln_phase(hx, (rb + l15) * NH, w_h, b_h, lnw_h, lnb_h,
                  s_sum_h, s_sq_h, wave, l15, khalf, acc);

    // x = ln(h2h) + ln(i2h)
#pragma unroll
    for (int t = 0; t < NT; ++t) {
        float4 lo = *(const float4*)&slot[t * 8 + 0];
        float4 hi = *(const float4*)&slot[t * 8 + 4];
        acc[t][0] += lo.x; acc[t][1] += lo.y; acc[t][2] += lo.z; acc[t][3] += lo.w;
        acc[t][4] += hi.x; acc[t][5] += hi.y; acc[t][6] += hi.z; acc[t][7] += hi.w;
    }

    // ---- gates + cx_new; accumulate cx layernorm stats (n = 1024) ----
    float go_k[4][8], cxn[4][8];
    float psum_c[8], psq_c[8];
#pragma unroll
    for (int v = 0; v < 8; ++v) { psum_c[v] = 0.f; psq_c[v] = 0.f; }

#pragma unroll
    for (int s = 0; s < 4; ++s) {
        const int h = wave * 64 + s * 16 + l15;
#pragma unroll
        for (int v = 0; v < 8; ++v) {
            const int r = v + khalf * 8;
            const float gi = acc[0  + s][v];
            const float gf = acc[4  + s][v];
            const float go = acc[8  + s][v];
            const float gc = acc[12 + s][v];
            const float cold = cx[(size_t)(rb + r) * NH + h];
            const float cn = sigmoid_f(gf + FGB) * cold + sigmoid_f(gi) * tanhf(gc);
            cxn[s][v]  = cn;
            go_k[s][v] = sigmoid_f(go);
            psum_c[v] += cn; psq_c[v] += cn * cn;
            out_c[(size_t)(rb + r) * NH + h] = cn;
        }
    }

#pragma unroll
    for (int v = 0; v < 8; ++v) {
#pragma unroll
        for (int off = 1; off < 16; off <<= 1) {
            psum_c[v] += __shfl_xor(psum_c[v], off, 32);
            psq_c[v]  += __shfl_xor(psq_c[v],  off, 32);
        }
        if (l15 == 0) {
            const int r = v + khalf * 8;
            atomicAdd(&s_sum_c[r], psum_c[v]);
            atomicAdd(&s_sq_c[r],  psq_c[v]);
        }
    }
    __syncthreads();

    // ---- hx_new = out_gate * tanh(LN(cx_new)) ----
    const float nH = 1024.0f;
#pragma unroll
    for (int s = 0; s < 4; ++s) {
        const int h = wave * 64 + s * 16 + l15;
        const float wo = lnw_o[h], bo = lnb_o[h];
#pragma unroll
        for (int v = 0; v < 8; ++v) {
            const int r = v + khalf * 8;
            const float su = s_sum_c[r], q = s_sq_c[r];
            const float m  = su / nH;
            const float var = fmaxf(0.f, (q - su * m) / (nH - 1.0f));
            const float inv = 1.0f / (sqrtf(var) + EPSV);
            const float lnv = (cxn[s][v] - m) * inv * wo + bo;
            out_h[(size_t)(rb + r) * NH + h] = go_k[s][v] * tanhf(lnv);
        }
    }
}

extern "C" void kernel_launch(void* const* d_in, const int* in_sizes, int n_in,
                              void* d_out, int out_size, void* d_ws, size_t ws_size,
                              hipStream_t stream) {
    (void)in_sizes; (void)n_in; (void)out_size; (void)d_ws; (void)ws_size;
    const float* xin   = (const float*)d_in[0];
    const float* hx    = (const float*)d_in[1];
    const float* cx    = (const float*)d_in[2];
    const float* w_i   = (const float*)d_in[3];
    const float* b_i   = (const float*)d_in[4];
    const float* w_h   = (const float*)d_in[5];
    const float* b_h   = (const float*)d_in[6];
    const float* lnw_i = (const float*)d_in[7];
    const float* lnb_i = (const float*)d_in[8];
    const float* lnw_h = (const float*)d_in[9];
    const float* lnb_h = (const float*)d_in[10];
    const float* lnw_o = (const float*)d_in[11];
    const float* lnb_o = (const float*)d_in[12];

    float* out_h = (float*)d_out;                      // hx_new first
    float* out_c = (float*)d_out + (size_t)NB * NH;    // then cx_new

    const size_t dynbytes = (size_t)512 * XI_STRIDE * sizeof(float);  // 270,336 B (< 320KB LDS)
    (void)hipFuncSetAttribute((const void*)lnlstm_fused_kernel,
                              hipFuncAttributeMaxDynamicSharedMemorySize,
                              (int)dynbytes);

    dim3 grid(NB / 16);
    dim3 block(512);
    lnlstm_fused_kernel<<<grid, block, dynbytes, stream>>>(
        xin, hx, cx, w_i, b_i, w_h, b_h,
        lnw_i, lnb_i, lnw_h, lnb_h, lnw_o, lnb_o,
        out_h, out_c);
}
